// MultiHeadAttention_85572928405803
// MI455X (gfx1250) — matrix operand order repaired and checked by reading, not compiled, not run on hardware
//
#include <hip/hip_runtime.h>
#include <hip/hip_bf16.h>
#include <stdint.h>

typedef _Float16 half_t;
typedef __attribute__((ext_vector_type(16))) _Float16 v16h;
typedef __attribute__((ext_vector_type(8)))  float    v8f;
typedef __attribute__((ext_vector_type(4)))  _Float16 half4;
typedef __attribute__((ext_vector_type(4)))  uint32_t u32x4;
typedef __attribute__((ext_vector_type(8)))  uint32_t u32x8;

#define B_  4
#define T_  2048
#define D_  1024
#define H_  16
#define DH_ 64

// ---------------------------------------------------------------------------
// gfx1250 async global->LDS copy (ASYNCcnt-tracked), ISA 15.18.3 op 97.
// ---------------------------------------------------------------------------
__device__ __forceinline__ void async_copy_b64(const half_t* g, half_t* l) {
  uint32_t loff =
      (uint32_t)(size_t)(__attribute__((address_space(3))) half_t*)l;
  asm volatile("global_load_async_to_lds_b64 %0, %1, off"
               :: "v"(loff), "v"((uint64_t)(size_t)g)
               : "memory");
}

__device__ __forceinline__ void wait_async_all() {
  asm volatile("s_wait_asynccnt 0x0" ::: "memory");
}

// ---------------------------------------------------------------------------
// gfx1250 Tensor Data Mover: 2-D f16 tile load, global -> LDS, with LDS row
// padding done by the D# (pad_interval = 32 DWORDs = 128B of data,
// pad_amount = 4 DWORDs = 16B  ->  matches LDS row stride 72 halves = 144B).
// D# packing per ISA 8.3/8.4; 2-operand form (VADDR2/3 = NULL, tensors <= 2D).
// Tracked by TENSORcnt.
// ---------------------------------------------------------------------------
__device__ __forceinline__ void tdm_load_tile_f16(
    const half_t* gbase, uint32_t lds_byte_addr,
    uint32_t rowlen_elems, uint32_t rowstride_elems, uint32_t nrows)
{
  uint64_t ga = (uint64_t)(size_t)gbase;
  u32x4 g0;
  g0[0] = 1u;                                   // count=1, user descriptor
  g0[1] = lds_byte_addr;                        // LDS dest (bytes)
  g0[2] = (uint32_t)ga;                         // global_addr[31:0]
  g0[3] = (uint32_t)((ga >> 32) & 0x1FFFFFFu)   // global_addr[56:32]
        | (2u << 30);                           // type = 2 ("image")
  u32x8 g1;
  g1[0] = (1u << 16)                            // data_size: 1 -> 2 bytes
        | (1u << 20)                            // pad_enable
        | (4u << 22)                            // pad_interval: 32 DWORDs
        | (3u << 25);                           // pad_amount:   4 DWORDs
  g1[1] = (rowlen_elems & 0xFFFFu) << 16;       // tensor_dim0[15:0]
  g1[2] = (nrows & 0xFFFFu) << 16;              // dim0[31:16]=0 | tensor_dim1[15:0]
  g1[3] = (rowlen_elems & 0xFFFFu) << 16;       // dim1[31:16]=0 | tile_dim0
  g1[4] = (nrows & 0xFFFFu);                    // tile_dim1 | tile_dim2=0
  g1[5] = rowstride_elems;                      // tensor_dim0_stride[31:0]
  g1[6] = 0u;                                   // stride hi | dim1_stride lo
  g1[7] = 0u;
  asm volatile("tensor_load_to_lds %0, %1" :: "s"(g0), "s"(g1) : "memory");
}

__device__ __forceinline__ void wait_tensor_all() {
  __builtin_amdgcn_s_wait_tensorcnt((short)0);
}

// ---------------------------------------------------------------------------
// Kernel 1: QKV projection.  C = X(BT x D) @ W(D x D) + b, z selects Q/K/V.
// Block = 256 threads (8 waves). M-tile 256, N-tile 64, K-step 32.
// f32->f16 conversion happens during staging (so VGPR-mediated copies here).
// Q pre-scaled by 1/sqrt(DH); K stored [b][h][t][d]; V stored [b][h][d][t].
// ---------------------------------------------------------------------------
__global__ __launch_bounds__(256) void qkv_proj_kernel(
    const float* __restrict__ x,
    const float* __restrict__ Wq, const float* __restrict__ Wk, const float* __restrict__ Wv,
    const float* __restrict__ bq, const float* __restrict__ bk, const float* __restrict__ bv,
    half_t* __restrict__ Qh, half_t* __restrict__ Kh, half_t* __restrict__ VTh)
{
  __shared__ half_t ldsA[256][40];   // [m][k]  (pad 32->40)
  __shared__ half_t ldsBT[64][40];   // [n][k]

  const int z       = blockIdx.z;
  const float* W    = (z == 0) ? Wq : (z == 1) ? Wk : Wv;
  const float* bias = (z == 0) ? bq : (z == 1) ? bk : bv;
  const int m0   = blockIdx.x * 256;
  const int n0   = blockIdx.y * 64;
  const int tid  = threadIdx.x;
  const int lane = tid & 31;
  const int wave = tid >> 5;
  const int lgrp = lane >> 4;
  const int l16  = lane & 15;

  v8f acc[2][4];
#pragma unroll
  for (int s = 0; s < 2; ++s)
#pragma unroll
    for (int nt = 0; nt < 4; ++nt)
#pragma unroll
      for (int e = 0; e < 8; ++e) acc[s][nt][e] = 0.f;

  for (int k0 = 0; k0 < D_; k0 += 32) {
#pragma unroll
    for (int i = 0; i < 8; ++i) {
      int f  = tid + i * 256;
      int r  = f >> 3;
      int c4 = (f & 7) << 2;
      float4 xv = *(const float4*)(x + (size_t)(m0 + r) * D_ + (k0 + c4));
      half4 hv = { (half_t)xv.x, (half_t)xv.y, (half_t)xv.z, (half_t)xv.w };
      *(half4*)&ldsA[r][c4] = hv;
    }
#pragma unroll
    for (int i = 0; i < 2; ++i) {
      int f  = tid + i * 256;
      int kk = f >> 4;
      int c4 = (f & 15) << 2;
      float4 wv = *(const float4*)(W + (size_t)(k0 + kk) * D_ + (n0 + c4));
      ldsBT[c4 + 0][kk] = (half_t)wv.x;
      ldsBT[c4 + 1][kk] = (half_t)wv.y;
      ldsBT[c4 + 2][kk] = (half_t)wv.z;
      ldsBT[c4 + 3][kk] = (half_t)wv.w;
    }
    __syncthreads();

    v16h bfrag[4];
#pragma unroll
    for (int nt = 0; nt < 4; ++nt) {
      const uint32_t* src = (const uint32_t*)&ldsBT[nt * 16 + l16][0];
      uint32_t* bb = (uint32_t*)&bfrag[nt];
#pragma unroll
      for (int v = 0; v < 8; ++v) bb[v] = src[4 * lgrp + (v & 3) + 8 * (v >> 2)];
    }
#pragma unroll
    for (int s = 0; s < 2; ++s) {
      int am = wave * 32 + s * 16 + l16;
      v16h afrag;
      const uint32_t* src = (const uint32_t*)&ldsA[am][0];
      uint32_t* aa = (uint32_t*)&afrag;
#pragma unroll
      for (int v = 0; v < 4; ++v) {
        aa[v]     = src[4 * lgrp + v];
        aa[v + 4] = src[4 * lgrp + 8 + v];
      }
#pragma unroll
      for (int nt = 0; nt < 4; ++nt)
        acc[s][nt] = __builtin_amdgcn_wmma_f32_16x16x32_f16(
            false, afrag, false, bfrag[nt], (short)0, acc[s][nt], false, false);
    }
    __syncthreads();
  }

  const float scale = (z == 0) ? 0.125f : 1.0f;
#pragma unroll
  for (int s = 0; s < 2; ++s)
#pragma unroll
    for (int nt = 0; nt < 4; ++nt)
#pragma unroll
      for (int r = 0; r < 8; ++r) {
        int gm = m0 + wave * 32 + s * 16 + 8 * lgrp + r;
        int gn = n0 + nt * 16 + l16;
        float vv = (acc[s][nt][r] + bias[gn]) * scale;
        half_t hv = (half_t)vv;
        int b = gm >> 11;
        int t = gm & (T_ - 1);
        int h = gn >> 6;
        int d = gn & (DH_ - 1);
        if (z == 0)      Qh[(((size_t)b * H_ + h) * T_ + t) * DH_ + d]  = hv;
        else if (z == 1) Kh[(((size_t)b * H_ + h) * T_ + t) * DH_ + d]  = hv;
        else             VTh[(((size_t)b * H_ + h) * DH_ + d) * T_ + t] = hv;
      }
}

// ---------------------------------------------------------------------------
// Kernel 2: flash attention per (b,h).  Block = 256 threads (8 waves),
// 128 query rows per block, KV tiles of 64, online softmax.
// K/V^T tiles are DOUBLE-BUFFERED and staged by the Tensor Data Mover:
// each wave issues two TDM descriptors (8 K-rows + 8 V^T-rows) per tile;
// the D# pads LDS rows to the 144B conflict-free stride.  TENSORcnt-waited.
// ---------------------------------------------------------------------------
__global__ __launch_bounds__(256) void flash_attn_kernel(
    const half_t* __restrict__ Qh, const half_t* __restrict__ Kh,
    const half_t* __restrict__ VTh, const unsigned char* __restrict__ mask,
    half_t* __restrict__ Oh)
{
  __shared__ half_t ldsK[2][64][72];   // [buf][t][d]  stride 144B
  __shared__ half_t ldsVT[2][64][72];  // [buf][d][t]
  __shared__ half_t ldsP[8][16][72];   // per-wave P tile [m][t]

  const int bh   = blockIdx.y;
  const int b    = bh >> 4;
  const int h    = bh & 15;
  const int q0   = blockIdx.x * 128;
  const int tid  = threadIdx.x;
  const int lane = tid & 31;
  const int wave = tid >> 5;
  const int lgrp = lane >> 4;
  const int l16  = lane & 15;
  const int wq   = q0 + wave * 16;

  const half_t* Kg = Kh  + (size_t)bh * T_ * DH_;   // [t][d]
  const half_t* Vg = VTh + (size_t)bh * DH_ * T_;   // [d][t]
  const uint32_t ldsK_base[2] = {
    (uint32_t)(size_t)(__attribute__((address_space(3))) half_t*)&ldsK[0][0][0],
    (uint32_t)(size_t)(__attribute__((address_space(3))) half_t*)&ldsK[1][0][0] };
  const uint32_t ldsV_base[2] = {
    (uint32_t)(size_t)(__attribute__((address_space(3))) half_t*)&ldsVT[0][0][0],
    (uint32_t)(size_t)(__attribute__((address_space(3))) half_t*)&ldsVT[1][0][0] };
  const uint32_t wrow = (uint32_t)(wave * 8);       // this wave's 8-row slice
  const uint32_t woff = wrow * 144u;                // byte offset in LDS tile

  // Q A-fragments resident in registers for whole KV loop
  v16h qfrag[2];
  {
    const half_t* qrow = Qh + (((size_t)bh) * T_ + wq + l16) * DH_;
#pragma unroll
    for (int c = 0; c < 2; ++c) {
      const uint32_t* src = (const uint32_t*)(qrow + 32 * c);
      uint32_t* aa = (uint32_t*)&qfrag[c];
#pragma unroll
      for (int v = 0; v < 4; ++v) {
        aa[v]     = src[4 * lgrp + v];
        aa[v + 4] = src[4 * lgrp + 8 + v];
      }
    }
  }

  float mrow[8], srow[8];
  v8f oacc[4];
#pragma unroll
  for (int r = 0; r < 8; ++r) { mrow[r] = -1e30f; srow[r] = 0.f; }
#pragma unroll
  for (int nt = 0; nt < 4; ++nt)
#pragma unroll
    for (int e = 0; e < 8; ++e) oacc[nt][e] = 0.f;

  // prologue: TDM-stage tile t0=0 into buffer 0 (each wave: 8 rows of K, V^T)
  tdm_load_tile_f16(Kg + (size_t)wrow * DH_, ldsK_base[0] + woff, DH_, DH_, 8);
  tdm_load_tile_f16(Vg + (size_t)wrow * T_,  ldsV_base[0] + woff, 64,  T_,  8);
  wait_tensor_all();
  __syncthreads();

  int buf = 0;
  for (int t0 = 0; t0 < T_; t0 += 64) {
    // TDM-prefetch next KV tile into the other buffer while we compute
    if (t0 + 64 < T_) {
      tdm_load_tile_f16(Kg + (size_t)(t0 + 64 + wrow) * DH_,
                        ldsK_base[buf ^ 1] + woff, DH_, DH_, 8);
      tdm_load_tile_f16(Vg + (size_t)wrow * T_ + (t0 + 64),
                        ldsV_base[buf ^ 1] + woff, 64, T_, 8);
    }

    // S = Q @ K^T  (16 x 64 per wave)
    v8f sacc[4];
#pragma unroll
    for (int nt = 0; nt < 4; ++nt)
#pragma unroll
      for (int e = 0; e < 8; ++e) sacc[nt][e] = 0.f;
#pragma unroll
    for (int c = 0; c < 2; ++c)
#pragma unroll
      for (int nt = 0; nt < 4; ++nt) {
        v16h kf;
        const uint32_t* src = (const uint32_t*)&ldsK[buf][nt * 16 + l16][0];
        uint32_t* bb = (uint32_t*)&kf;
#pragma unroll
        for (int v = 0; v < 8; ++v) bb[v] = src[16 * c + 4 * lgrp + (v & 3) + 8 * (v >> 2)];
        sacc[nt] = __builtin_amdgcn_wmma_f32_16x16x32_f16(
            false, qfrag[c], false, kf, (short)0, sacc[nt], false, false);
      }

    // additive mask + online softmax
#pragma unroll
    for (int r = 0; r < 8; ++r) {
      int qrow = wq + 8 * lgrp + r;
#pragma unroll
      for (int nt = 0; nt < 4; ++nt) {
        int kcol = t0 + nt * 16 + l16;
        float s = sacc[nt][r];
        if (!mask[(size_t)qrow * T_ + kcol]) s += -1e9f;
        sacc[nt][r] = s;
      }
      float rm = fmaxf(fmaxf(sacc[0][r], sacc[1][r]), fmaxf(sacc[2][r], sacc[3][r]));
#pragma unroll
      for (int off = 1; off < 16; off <<= 1)
        rm = fmaxf(rm, __shfl_xor(rm, off, 32));
      float mnew  = fmaxf(mrow[r], rm);
      float alpha = __expf(mrow[r] - mnew);
      mrow[r] = mnew;
      float psum = 0.f;
#pragma unroll
      for (int nt = 0; nt < 4; ++nt) {
        float p = __expf(sacc[nt][r] - mnew);
        psum += p;
        ldsP[wave][8 * lgrp + r][nt * 16 + l16] = (half_t)p;
      }
#pragma unroll
      for (int off = 1; off < 16; off <<= 1)
        psum += __shfl_xor(psum, off, 32);
      srow[r] = srow[r] * alpha + psum;
#pragma unroll
      for (int nt = 0; nt < 4; ++nt) oacc[nt][r] *= alpha;
    }

    // O += P @ V
#pragma unroll
    for (int c = 0; c < 2; ++c) {
      v16h pf;
      const uint32_t* src = (const uint32_t*)&ldsP[wave][l16][0];
      uint32_t* aa = (uint32_t*)&pf;
#pragma unroll
      for (int v = 0; v < 4; ++v) {
        aa[v]     = src[16 * c + 4 * lgrp + v];
        aa[v + 4] = src[16 * c + 4 * lgrp + 8 + v];
      }
#pragma unroll
      for (int nt = 0; nt < 4; ++nt) {
        v16h vf;
        const uint32_t* vsrc = (const uint32_t*)&ldsVT[buf][nt * 16 + l16][0];
        uint32_t* bb = (uint32_t*)&vf;
#pragma unroll
        for (int v = 0; v < 8; ++v) bb[v] = vsrc[16 * c + 4 * lgrp + (v & 3) + 8 * (v >> 2)];
        oacc[nt] = __builtin_amdgcn_wmma_f32_16x16x32_f16(
            false, pf, false, vf, (short)0, oacc[nt], false, false);
      }
    }

    wait_tensor_all();    // our prefetch for the next tile has landed
    __syncthreads();      // everyone done reading current buffers
    buf ^= 1;
  }

  // normalize, merge heads, store f16 O[b][t][h*DH+d]
#pragma unroll
  for (int nt = 0; nt < 4; ++nt)
#pragma unroll
    for (int r = 0; r < 8; ++r) {
      int qrow = wq + 8 * lgrp + r;
      int dcol = h * DH_ + nt * 16 + l16;
      float o = oacc[nt][r] / srow[r];
      Oh[((size_t)b * T_ + qrow) * D_ + dcol] = (half_t)o;
    }
}

// ---------------------------------------------------------------------------
// Kernel 3: output projection  out = O(f16) @ Wo + bo  -> fp32
// A tile (already f16) staged with async global->LDS copies (ASYNCcnt),
// overlapped with the Wo f32->f16 conversion staging.
// ---------------------------------------------------------------------------
__global__ __launch_bounds__(256) void out_proj_kernel(
    const half_t* __restrict__ Oh, const float* __restrict__ Wo,
    const float* __restrict__ bo, float* __restrict__ out)
{
  __shared__ half_t ldsA[256][40];
  __shared__ half_t ldsBT[64][40];

  const int m0   = blockIdx.x * 256;
  const int n0   = blockIdx.y * 64;
  const int tid  = threadIdx.x;
  const int lane = tid & 31;
  const int wave = tid >> 5;
  const int lgrp = lane >> 4;
  const int l16  = lane & 15;

  v8f acc[2][4];
#pragma unroll
  for (int s = 0; s < 2; ++s)
#pragma unroll
    for (int nt = 0; nt < 4; ++nt)
#pragma unroll
      for (int e = 0; e < 8; ++e) acc[s][nt][e] = 0.f;

  for (int k0 = 0; k0 < D_; k0 += 32) {
#pragma unroll
    for (int i = 0; i < 8; ++i) {
      int f = tid + i * 256;
      int r = f >> 3;
      int c = (f & 7) << 2;
      async_copy_b64(Oh + (size_t)(m0 + r) * D_ + k0 + c, &ldsA[r][c]);
    }
#pragma unroll
    for (int i = 0; i < 2; ++i) {
      int f  = tid + i * 256;
      int kk = f >> 4;
      int c4 = (f & 15) << 2;
      float4 wv = *(const float4*)(Wo + (size_t)(k0 + kk) * D_ + (n0 + c4));
      ldsBT[c4 + 0][kk] = (half_t)wv.x;
      ldsBT[c4 + 1][kk] = (half_t)wv.y;
      ldsBT[c4 + 2][kk] = (half_t)wv.z;
      ldsBT[c4 + 3][kk] = (half_t)wv.w;
    }
    wait_async_all();
    __syncthreads();

    v16h bfrag[4];
#pragma unroll
    for (int nt = 0; nt < 4; ++nt) {
      const uint32_t* src = (const uint32_t*)&ldsBT[nt * 16 + l16][0];
      uint32_t* bb = (uint32_t*)&bfrag[nt];
#pragma unroll
      for (int v = 0; v < 8; ++v) bb[v] = src[4 * lgrp + (v & 3) + 8 * (v >> 2)];
    }
#pragma unroll
    for (int s = 0; s < 2; ++s) {
      int am = wave * 32 + s * 16 + l16;
      v16h afrag;
      const uint32_t* src = (const uint32_t*)&ldsA[am][0];
      uint32_t* aa = (uint32_t*)&afrag;
#pragma unroll
      for (int v = 0; v < 4; ++v) {
        aa[v]     = src[4 * lgrp + v];
        aa[v + 4] = src[4 * lgrp + 8 + v];
      }
#pragma unroll
      for (int nt = 0; nt < 4; ++nt)
        acc[s][nt] = __builtin_amdgcn_wmma_f32_16x16x32_f16(
            false, afrag, false, bfrag[nt], (short)0, acc[s][nt], false, false);
    }
    __syncthreads();
  }

#pragma unroll
  for (int s = 0; s < 2; ++s)
#pragma unroll
    for (int nt = 0; nt < 4; ++nt)
#pragma unroll
      for (int r = 0; r < 8; ++r) {
        int gm = m0 + wave * 32 + s * 16 + 8 * lgrp + r;
        int gn = n0 + nt * 16 + l16;
        out[(size_t)gm * D_ + gn] = acc[s][nt][r] + bo[gn];
      }
}

// ---------------------------------------------------------------------------
extern "C" void kernel_launch(void* const* d_in, const int* in_sizes, int n_in,
                              void* d_out, int out_size, void* d_ws, size_t ws_size,
                              hipStream_t stream) {
  const float* x            = (const float*)d_in[0];
  const unsigned char* mask = (const unsigned char*)d_in[1];
  const float* Wq = (const float*)d_in[2];
  const float* bq = (const float*)d_in[3];
  const float* Wk = (const float*)d_in[4];
  const float* bk = (const float*)d_in[5];
  const float* Wv = (const float*)d_in[6];
  const float* bv = (const float*)d_in[7];
  const float* Wo = (const float*)d_in[8];
  const float* bo = (const float*)d_in[9];
  float* out = (float*)d_out;

  const size_t BTD = (size_t)B_ * T_ * D_;
  half_t* Qh  = (half_t*)d_ws;     // 16 MB each; 64 MB total workspace used
  half_t* Kh  = Qh  + BTD;
  half_t* VTh = Kh  + BTD;
  half_t* Oh  = VTh + BTD;

  dim3 blk(256);
  qkv_proj_kernel<<<dim3((B_ * T_) / 256, D_ / 64, 3), blk, 0, stream>>>(
      x, Wq, Wk, Wv, bq, bk, bv, Qh, Kh, VTh);
  flash_attn_kernel<<<dim3(T_ / 128, B_ * H_), blk, 0, stream>>>(
      Qh, Kh, VTh, mask, Oh);
  out_proj_kernel<<<dim3((B_ * T_) / 256, D_ / 64), blk, 0, stream>>>(
      Oh, Wo, bo, out);
}
